// SelfAttention_15865609191868
// MI455X (gfx1250) — matrix operand order repaired and checked by reading, not compiled, not run on hardware
//
#include <hip/hip_runtime.h>
#include <hip/hip_bf16.h>

typedef __attribute__((ext_vector_type(16))) _Float16 v16h;
typedef __attribute__((ext_vector_type(8)))  _Float16 v8h;
typedef __attribute__((ext_vector_type(8)))  float    v8f;

union V16 { v16h v; v8h h[2]; };

#define NB   2
#define SEQ  2048
#define NH   16
#define HD   64
#define QT   16
#define KT   32
#define NQT  (SEQ/QT)
#define L2E  1.44269504088896340736f
#define PSTRIDE 40                     // f16 elems per P row: 80B, 16B-aligned

// ---- single-instruction DPP-fused reductions over a 16-lane row ------------
// d = op(perm(x), x): one VALU op per butterfly step, no canonicalize movs.
#define DPP_STEP(NAME, OP, CTRL)                                   \
  __device__ __forceinline__ float NAME(float x) {                 \
    float d;                                                       \
    asm(OP " %0, %1, %2 " CTRL                                     \
        : "=v"(d) : "v"(x), "v"(x));                               \
    return d;                                                      \
  }
DPP_STEP(max_x1, "v_max_num_f32_dpp", "quad_perm:[1,0,3,2] row_mask:0xf bank_mask:0xf bound_ctrl:1")
DPP_STEP(max_x2, "v_max_num_f32_dpp", "quad_perm:[2,3,0,1] row_mask:0xf bank_mask:0xf bound_ctrl:1")
DPP_STEP(max_hm, "v_max_num_f32_dpp", "row_half_mirror row_mask:0xf bank_mask:0xf bound_ctrl:1")
DPP_STEP(max_rm, "v_max_num_f32_dpp", "row_mirror row_mask:0xf bank_mask:0xf bound_ctrl:1")
DPP_STEP(add_x1, "v_add_f32_dpp",     "quad_perm:[1,0,3,2] row_mask:0xf bank_mask:0xf bound_ctrl:1")
DPP_STEP(add_x2, "v_add_f32_dpp",     "quad_perm:[2,3,0,1] row_mask:0xf bank_mask:0xf bound_ctrl:1")
DPP_STEP(add_hm, "v_add_f32_dpp",     "row_half_mirror row_mask:0xf bank_mask:0xf bound_ctrl:1")
DPP_STEP(add_rm, "v_add_f32_dpp",     "row_mirror row_mask:0xf bank_mask:0xf bound_ctrl:1")

__device__ __forceinline__ float fmax_raw(float a, float b) {
  float d;
  asm("v_max_num_f32 %0, %1, %2" : "=v"(d) : "v"(a), "v"(b));
  return d;
}

// ---------------------------------------------------------------------------
// Kernel 1: repack fp32 qkv[b][s][3][h][d] into f16 workspace:
//   Qh [b][h][s][d] (pre-scaled by 0.125), Kh [b][h][s][d], Vt [b][h][d][s]
// ---------------------------------------------------------------------------
__global__ __launch_bounds__(256) void cvt_qkv(const float* __restrict__ qkv,
                                               _Float16* __restrict__ ws) {
  const int N = NB * NH * SEQ * HD;
  int t = blockIdx.x * 256 + threadIdx.x;
  if (t >= N) return;
  int which = blockIdx.y;           // 0=Q, 1=K, 2=V(transposed)
  if (which < 2) {
    int d = t % HD;
    int s = (t / HD) % SEQ;
    int h = (t / (HD * SEQ)) % NH;
    int b = t / (HD * SEQ * NH);
    float v = qkv[(((size_t)(b * SEQ + s) * 3 + which) * NH + h) * HD + d];
    if (which == 0) v *= 0.125f;    // fold softmax scale into Q
    ws[(size_t)which * N + t] = (_Float16)v;
  } else {
    int s = t % SEQ;
    int d = (t / SEQ) % HD;
    int h = (t / (SEQ * HD)) % NH;
    int b = t / (SEQ * HD * NH);
    float v = qkv[(((size_t)(b * SEQ + s) * 3 + 2) * NH + h) * HD + d];
    ws[(size_t)2 * N + t] = (_Float16)v;
  }
}

// ---------------------------------------------------------------------------
// Kernel 2: causal flash attention, one wave32 per 16-query tile.
// ---------------------------------------------------------------------------
__global__ __launch_bounds__(32) void flash_fwd(const _Float16* __restrict__ ws,
                                                float* __restrict__ out) {
  const int N = NB * NH * SEQ * HD;
  const _Float16* Qh = ws;
  const _Float16* Kh = ws + (size_t)N;
  const _Float16* Vt = ws + (size_t)2 * N;

  const int blk  = blockIdx.x;
  const int qt   = blk % NQT;          // query tile index
  const int bh   = blk / NQT;          // b*NH + h
  const int lane = threadIdx.x & 31;
  const int n    = lane & 15;          // column within 16-lane group
  const int hx   = lane >> 4;          // lane half (selects M+8 / K-range)

  __shared__ _Float16 Ph[16 * PSTRIDE];  // P tile, f16, padded rows

  const _Float16* Qp = Qh + ((size_t)bh * SEQ + qt * QT) * HD;
  const _Float16* Kp = Kh + (size_t)bh * SEQ * HD;
  const _Float16* Vp = Vt + (size_t)bh * HD * SEQ;

  // ---- Q tile in WMMA A layout (row m = lane%16, 16 consecutive d per chunk)
  V16 qa[2];
#pragma unroll
  for (int c = 0; c < 2; ++c) {
    qa[c].h[0] = *(const v8h*)(Qp + n * HD + 32 * c + 16 * hx);
    qa[c].h[1] = *(const v8h*)(Qp + n * HD + 32 * c + 16 * hx + 8);
  }

  float mrow[8], lrow[8];
  v8f oacc[4];
#pragma unroll
  for (int i = 0; i < 8; ++i) { mrow[i] = -1e30f; lrow[i] = 0.0f; }
#pragma unroll
  for (int c = 0; c < 4; ++c) oacc[c] = (v8f){};

  const int njt = qt / 2 + 1;          // # of 32-key tiles up to the diagonal
  for (int j = 0; j < njt; ++j) {
    const int ks = j * KT;

    // ---- K tile in B layout: lane = key column, 16 consecutive d per v16h
    V16 kb[2][2];
#pragma unroll
    for (int nh2 = 0; nh2 < 2; ++nh2) {
      const _Float16* kr = Kp + (size_t)(ks + 16 * nh2 + n) * HD;
#pragma unroll
      for (int c = 0; c < 2; ++c) {
        kb[nh2][c].h[0] = *(const v8h*)(kr + 32 * c + 16 * hx);
        kb[nh2][c].h[1] = *(const v8h*)(kr + 32 * c + 16 * hx + 8);
      }
    }

    // ---- prefetch next k-tile (one 128B K row / two 64B V rows per lane)
    if (j + 1 < njt) {
      __builtin_prefetch(Kp + (size_t)(ks + KT + lane) * HD, 0, 3);
      __builtin_prefetch(Vp + (size_t)lane        * SEQ + ks + KT, 0, 3);
      __builtin_prefetch(Vp + (size_t)(lane + 32) * SEQ + ks + KT, 0, 3);
    }

    // ---- S = Q * K^T  (16x32 scores = two 16x16 C tiles)
    v8f s0 = (v8f){}, s1 = (v8f){};
    s0 = __builtin_amdgcn_wmma_f32_16x16x32_f16(false, qa[0].v, false, kb[0][0].v, (short)0, s0, false, false);
    s0 = __builtin_amdgcn_wmma_f32_16x16x32_f16(false, qa[1].v, false, kb[0][1].v, (short)0, s0, false, false);
    s1 = __builtin_amdgcn_wmma_f32_16x16x32_f16(false, qa[0].v, false, kb[1][0].v, (short)0, s1, false, false);
    s1 = __builtin_amdgcn_wmma_f32_16x16x32_f16(false, qa[1].v, false, kb[1][1].v, (short)0, s1, false, false);

    // ---- causal mask (only the diagonal tile ever needs it)
    if (j == njt - 1) {
#pragma unroll
      for (int i = 0; i < 8; ++i) {
        int q = qt * QT + i + 8 * hx;
        s0[i] = (ks + n      > q) ? s0[i] - 10000.0f : s0[i];
        s1[i] = (ks + 16 + n > q) ? s1[i] - 10000.0f : s1[i];
      }
    }

    // ---- online softmax, stage-at-a-time so all 8 row chains interleave ----
    float t[8], alpha[8], p0[8], p1[8], r[8];
#pragma unroll
    for (int i = 0; i < 8; ++i) t[i] = fmax_raw(s0[i], s1[i]);
#pragma unroll
    for (int i = 0; i < 8; ++i) t[i] = max_x1(t[i]);
#pragma unroll
    for (int i = 0; i < 8; ++i) t[i] = max_x2(t[i]);
#pragma unroll
    for (int i = 0; i < 8; ++i) t[i] = max_hm(t[i]);
#pragma unroll
    for (int i = 0; i < 8; ++i) t[i] = max_rm(t[i]);
#pragma unroll
    for (int i = 0; i < 8; ++i) t[i] = fmax_raw(mrow[i], t[i]);   // m_new
#pragma unroll
    for (int i = 0; i < 8; ++i) {
      float msc = t[i] * L2E;                      // exp2(s*L2E - m*L2E)
      alpha[i] = __builtin_amdgcn_exp2f(fmaf(mrow[i], L2E, -msc));
      p0[i]    = __builtin_amdgcn_exp2f(fmaf(s0[i],  L2E, -msc));
      p1[i]    = __builtin_amdgcn_exp2f(fmaf(s1[i],  L2E, -msc));
      mrow[i]  = t[i];
    }
#pragma unroll
    for (int i = 0; i < 8; ++i) r[i] = p0[i] + p1[i];
#pragma unroll
    for (int i = 0; i < 8; ++i) r[i] = add_x1(r[i]);
#pragma unroll
    for (int i = 0; i < 8; ++i) r[i] = add_x2(r[i]);
#pragma unroll
    for (int i = 0; i < 8; ++i) r[i] = add_hm(r[i]);
#pragma unroll
    for (int i = 0; i < 8; ++i) r[i] = add_rm(r[i]);
#pragma unroll
    for (int i = 0; i < 8; ++i) lrow[i] = fmaf(lrow[i], alpha[i], r[i]);
#pragma unroll
    for (int c = 0; c < 4; ++c)
#pragma unroll
      for (int i = 0; i < 8; ++i) oacc[c][i] *= alpha[i];

    // ---- C-layout P -> LDS (f16) -> A-layout reload (2x ds_load_b128)
    __syncthreads();
#pragma unroll
    for (int i = 0; i < 8; ++i) {
      Ph[(i + 8 * hx) * PSTRIDE + n]      = (_Float16)p0[i];
      Ph[(i + 8 * hx) * PSTRIDE + n + 16] = (_Float16)p1[i];
    }
    __syncthreads();
    V16 pa;
    pa.h[0] = *(const v8h*)(&Ph[n * PSTRIDE + 16 * hx]);       // K = 16hx..16hx+7
    pa.h[1] = *(const v8h*)(&Ph[n * PSTRIDE + 16 * hx + 8]);   // K = 16hx+8..
    __syncthreads();

    // ---- O += P * V  (V^T layout makes B-operand loads contiguous)
#pragma unroll
    for (int c = 0; c < 4; ++c) {
      V16 vb;
      const _Float16* vr = Vp + (size_t)(16 * c + n) * SEQ + ks + 16 * hx;
      vb.h[0] = *(const v8h*)(vr);
      vb.h[1] = *(const v8h*)(vr + 8);
      oacc[c] = __builtin_amdgcn_wmma_f32_16x16x32_f16(false, pa.v, false, vb.v, (short)0, oacc[c], false, false);
    }
  }

  // ---- epilogue: normalize by l (fast rcp), store fp32 out[b][q][h][d]
  float inv[8];
#pragma unroll
  for (int i = 0; i < 8; ++i) inv[i] = __builtin_amdgcn_rcpf(lrow[i]);
  const int b = bh / NH, h = bh % NH;
#pragma unroll
  for (int c = 0; c < 4; ++c)
#pragma unroll
    for (int i = 0; i < 8; ++i) {
      int q = qt * QT + i + 8 * hx;
      int d = 16 * c + n;
      out[(((size_t)b * SEQ + q) * NH + h) * HD + d] = oacc[c][i] * inv[i];
    }
}

extern "C" void kernel_launch(void* const* d_in, const int* in_sizes, int n_in,
                              void* d_out, int out_size, void* d_ws, size_t ws_size,
                              hipStream_t stream) {
  const float* qkv = (const float*)d_in[0];
  _Float16*    ws  = (_Float16*)d_ws;     // needs 3 * 2*16*2048*64 * 2B = 24 MiB
  float*       out = (float*)d_out;
  const int N = NB * NH * SEQ * HD;
  dim3 cgrid((N + 255) / 256, 3);
  cvt_qkv<<<cgrid, 256, 0, stream>>>(qkv, ws);
  flash_fwd<<<NB * NH * NQT, 32, 0, stream>>>(ws, out);
}
